// SMN_73435350827621
// MI455X (gfx1250) — matrix operand order repaired and checked by reading, not compiled, not run on hardware
//
#include <hip/hip_runtime.h>

typedef _Float16 f16;
typedef __attribute__((ext_vector_type(16))) _Float16 v16h;
typedef __attribute__((ext_vector_type(8)))  _Float16 v8h;
typedef __attribute__((ext_vector_type(8)))  float    v8f;

#define DEV __device__ __forceinline__

// Hardware tanh (CDNA5 TRANS32 op). Prefer the builtin (compiler handles the
// TRANS co-execution hazard); fall back to inline asm + v_nop per ISA 15.5.
DEV float tanh_hw(float x) {
#if __has_builtin(__builtin_amdgcn_tanhf)
  return __builtin_amdgcn_tanhf(x);
#else
  float y;
  asm("v_tanh_f32 %0, %1\n\tv_nop" : "=v"(y) : "v"(x));
  return y;
#endif
}

DEV float gelu_f(float x) {
  const float c = 0.7978845608028654f; // sqrt(2/pi)
  float inner = c * __builtin_fmaf(0.044715f * x * x, x, x);
  return 0.5f * x * (1.f + tanh_hw(inner));
}

// A fragment (16x32 f16) gathered from row-major LDS tile [16][lda] (halfs).
// Layout per ISA 7.12.2: lanes 0-15 -> M=lane; VGPR p<4: K=lh*8+2p+{0,1};
// p>=4: K=16+lh*8+2(p-4)+{0,1}.
DEV v16h load_a_lds(const f16* src, int lda, int k0, int lane) {
  int m = lane & 15, lh = lane >> 4;
  v16h a;
#pragma unroll
  for (int p = 0; p < 8; ++p) {
    int kb = k0 + ((p >> 2) << 4) + (lh << 3) + ((p & 3) << 1);
    const f16* q = src + m * lda + kb;
    a[2 * p]     = q[0];
    a[2 * p + 1] = q[1];
  }
  return a;
}

// B fragment (32x16 f16) from N-major (pre-transposed) weight wt[N][ldk]:
// lane holds column n = n0+(lane&15), K = lh*8 + e (e<8), 16 + lh*8 + (e-8) (e>=8) -> 2 b128 loads.
DEV v16h load_b_glb(const f16* wt, int ldk, int n0, int k0, int lane) {
  int n = n0 + (lane & 15), lh = lane >> 4;
  const v8h* q = (const v8h*)(wt + n * ldk + k0 + (lh << 3));
  v8h b0 = q[0], b1 = q[2];
  v16h b;
#pragma unroll
  for (int e = 0; e < 8; ++e) { b[e] = b0[e]; b[8 + e] = b1[e]; }
  return b;
}

DEV v8f wmma_f16(v16h a, v16h b, v8f c) {
  return __builtin_amdgcn_wmma_f32_16x16x32_f16(false, a, false, b, (short)0, c,
                                                false, false);
}

// ---------------------------------------------------------------- prep ------
__global__ void transpose_f16_kernel(const float* __restrict__ w,
                                     f16* __restrict__ wt, int K, int N) {
  int idx = blockIdx.x * blockDim.x + threadIdx.x;
  if (idx >= K * N) return;
  int n = idx / K, k = idx - n * K;
  wt[idx] = (f16)w[k * N + n];  // wt[n][k] = w[k][n]
}

// ---------------------------------------------------------------- proj ------
// h[s] = gelu(x[s] @ W_proj), x (32,8,64), W_proj^T in f16 (64x64 N-major)
__global__ void __launch_bounds__(128)
proj_kernel(const float* __restrict__ x, const f16* __restrict__ wt_proj,
            float* __restrict__ h) {
  __shared__ f16 ah[16][64];
  const int tid = threadIdx.x, lane = tid & 31, wave = tid >> 5;
  const int s = blockIdx.x;
  for (int idx = tid; idx < 16 * 64; idx += 128) {
    int r = idx >> 6, d = idx & 63;
    ah[r][d] = (f16)((r < 8) ? x[(s * 8 + r) * 64 + d] : 0.f);
  }
  __syncthreads();
  int nt = wave;  // 4 waves cover N=64
  v8f c = {};
#pragma unroll
  for (int k0 = 0; k0 < 64; k0 += 32)
    c = wmma_f16(load_a_lds(&ah[0][0], 64, k0, lane),
                 load_b_glb(wt_proj, 64, nt * 16, k0, lane), c);
  int ncol = nt * 16 + (lane & 15), mh = (lane >> 4) * 8;
#pragma unroll
  for (int r = 0; r < 8; ++r) {
    int m = mh + r;
    if (m < 8) h[(s * 8 + m) * 64 + ncol] = gelu_f(c[r]);
  }
}

// --------------------------------------------------------------- mixer ------
template <int NROWS, int NSTORE, bool CROSS>
__global__ void __launch_bounds__(128)
mixer_kernel(const float* __restrict__ in_rows,  // (32,8,64)
             const float* __restrict__ ln_g, const float* __restrict__ ln_b,
             const f16* __restrict__ wt_p1w1, const float* __restrict__ p1b1,
             const f16* __restrict__ wt_p1w2, const float* __restrict__ p1b2,
             const f16* __restrict__ wt_p2w1, const float* __restrict__ p2b1,
             const float* __restrict__ p2w2, const float* __restrict__ p2b2,
             const float* __restrict__ w3, const float* __restrict__ b3,
             float* __restrict__ out) {
  __shared__ float xin[16][64];
  __shared__ float maskrow[16];
  __shared__ f16   xnh[16][64];
  __shared__ f16   xh[16][64];
  __shared__ f16   t1h[16][128];
  __shared__ float phi1f[16][64];
  __shared__ f16   phi1h[16][64];
  __shared__ float h1m[16][128];   // holds h1 + b1 (bias folded in epilogue)
  __shared__ float h2m[16][128];
  __shared__ float phi2m[16][16][2];
  __shared__ float sw2[128][2];

  const int tid = threadIdx.x, lane = tid & 31, wave = tid >> 5;
  const int s = blockIdx.x, t = CROSS ? blockIdx.y : 0;

  __builtin_prefetch(wt_p1w1, 0, 0);   // global_prefetch_b8
  __builtin_prefetch(wt_p2w1, 0, 0);

  for (int idx = tid; idx < 16 * 64; idx += 128) {
    int r = idx >> 6, d = idx & 63;
    float v = 0.f;
    if (r < NROWS) {
      int src = CROSS ? ((r < 8) ? (s * 8 + r) : (t * 8 + (r - 8)))
                      : (s * 8 + r);
      v = in_rows[src * 64 + d];
    }
    xin[r][d] = v;
  }
  for (int idx = tid; idx < 256; idx += 128) (&sw2[0][0])[idx] = p2w2[idx];
  __syncthreads();

  // LayerNorm + row mask (wave handles a row; 2 elems/lane, shfl reduce)
  for (int r = wave; r < 16; r += 4) {
    float a0 = xin[r][lane], a1 = xin[r][lane + 32];
    float s1 = a0 + a1, s2 = a0 * a0 + a1 * a1;
    float nz = (a0 != 0.f || a1 != 0.f) ? 1.f : 0.f;
#pragma unroll
    for (int off = 16; off > 0; off >>= 1) {
      s1 += __shfl_xor(s1, off, 32);
      s2 += __shfl_xor(s2, off, 32);
      nz += __shfl_xor(nz, off, 32);
    }
    float mk   = (r < NROWS && nz > 0.f) ? 1.f : 0.f;
    float mean = s1 * (1.f / 64.f);
    float var  = s2 * (1.f / 64.f) - mean * mean;
    float inv  = rsqrtf(var + 1e-3f);
    xnh[r][lane]      = (f16)(((a0 - mean) * inv * ln_g[lane] + ln_b[lane]) * mk);
    xnh[r][lane + 32] = (f16)(((a1 - mean) * inv * ln_g[lane + 32] + ln_b[lane + 32]) * mk);
    xh[r][lane]       = (f16)a0;
    xh[r][lane + 32]  = (f16)a1;
    if (lane == 0) maskrow[r] = mk;
  }
  __syncthreads();

  // t1 = gelu(xn @ phi1.W1 + b1) : 16x64 @ 64x128 -> 8 N-tiles
  for (int nt = wave; nt < 8; nt += 4) {
    v8f c = {};
#pragma unroll
    for (int k0 = 0; k0 < 64; k0 += 32)
      c = wmma_f16(load_a_lds(&xnh[0][0], 64, k0, lane),
                   load_b_glb(wt_p1w1, 64, nt * 16, k0, lane), c);
    int ncol = nt * 16 + (lane & 15), mh = (lane >> 4) * 8;
    float bv = p1b1[ncol];
#pragma unroll
    for (int r = 0; r < 8; ++r) t1h[mh + r][ncol] = (f16)gelu_f(c[r] + bv);
  }
  __syncthreads();

  // phi1 = (t1 @ phi1.W2 + b2) * mask : 16x128 @ 128x64 -> 4 N-tiles
  for (int nt = wave; nt < 4; nt += 4) {
    v8f c = {};
#pragma unroll
    for (int k0 = 0; k0 < 128; k0 += 32)
      c = wmma_f16(load_a_lds(&t1h[0][0], 128, k0, lane),
                   load_b_glb(wt_p1w2, 128, nt * 16, k0, lane), c);
    int ncol = nt * 16 + (lane & 15), mh = (lane >> 4) * 8;
    float bv = p1b2[ncol];
#pragma unroll
    for (int r = 0; r < 8; ++r) {
      int m = mh + r;
      float v = (c[r] + bv) * maskrow[m];
      phi1f[m][ncol] = v;
      phi1h[m][ncol] = (f16)v;
    }
  }
  __syncthreads();

  // h1 = phi1 @ phi2.W1 + b1 ; h2 = x_orig @ phi2.W1 : 16 tile-jobs / 4 waves
  for (int job = wave; job < 16; job += 4) {
    const bool isH1 = job < 8;
    const int  nt   = isH1 ? job : job - 8;
    const f16* A    = isH1 ? &phi1h[0][0] : &xh[0][0];
    float*     dst  = isH1 ? &h1m[0][0] : &h2m[0][0];
    v8f c = {};
#pragma unroll
    for (int k0 = 0; k0 < 64; k0 += 32)
      c = wmma_f16(load_a_lds(A, 64, k0, lane),
                   load_b_glb(wt_p2w1, 64, nt * 16, k0, lane), c);
    int ncol = nt * 16 + (lane & 15), mh = (lane >> 4) * 8;
    float bv = isH1 ? p2b1[ncol] : 0.f;   // fold phi2.b1 into h1
#pragma unroll
    for (int r = 0; r < 8; ++r) dst[(mh + r) * 128 + ncol] = c[r] + bv;
  }
  __syncthreads();

  // phi2[i][j][:] = gelu(h1[i]-h2[j]) @ W2 + b2   (hid never materialized)
  for (int pr = tid; pr < NROWS * NROWS; pr += 128) {
    int i = pr / NROWS, j = pr % NROWS;
    float a0 = 0.f, a1 = 0.f;
    for (int c = 0; c < 128; ++c) {
      float v = gelu_f(h1m[i][c] - h2m[j][c]);
      a0 = __builtin_fmaf(v, sw2[c][0], a0);
      a1 = __builtin_fmaf(v, sw2[c][1], a1);
    }
    phi2m[i][j][0] = a0 + p2b2[0];
    phi2m[i][j][1] = a1 + p2b2[1];
  }
  __syncthreads();

  // mix[i,d,h] = sum_j phi1[j,d]*phi2[i,j,h]; out = (gelu(mix)@W3+b3)+x, *mask
  const int obase = (CROSS ? (s * 32 + t) : s) * (NSTORE * 64);
  for (int idx = tid; idx < NSTORE * 64; idx += 128) {
    int i = idx >> 6, d = idx & 63;
    float m0 = 0.f, m1 = 0.f;
#pragma unroll
    for (int j = 0; j < NROWS; ++j) {
      float p = phi1f[j][d];
      m0 = __builtin_fmaf(p, phi2m[i][j][0], m0);
      m1 = __builtin_fmaf(p, phi2m[i][j][1], m1);
    }
    float o = gelu_f(m0) * w3[0] + gelu_f(m1) * w3[1] + b3[0];
    out[obase + idx] = (o + xin[i][d]) * maskrow[i];
  }
}

// --------------------------------------------------------------- final ------
__global__ void __launch_bounds__(128)
final_kernel(const float* __restrict__ xc,  // (32,32,8,64)
             const float* __restrict__ ln_g, const float* __restrict__ ln_b,
             const f16* __restrict__ wt_p1w1, const float* __restrict__ p1b1,
             const f16* __restrict__ wt_p1w2, const float* __restrict__ p1b2,
             const f16* __restrict__ wt_p2w1, const float* __restrict__ p2b1,
             const float* __restrict__ p2w2, const float* __restrict__ p2b2,
             const float* __restrict__ w3, const float* __restrict__ b3,
             float* __restrict__ xrep) {  // (32,32,64)
  __shared__ float xin[8][64];
  __shared__ float maskrow[16];
  __shared__ f16   xnh[16][64];
  __shared__ f16   t1h[16][128];
  __shared__ f16   t2h[16][128];
  __shared__ float phi1f[16][64];
  __shared__ float ph2[8][2];

  const int tid = threadIdx.x, lane = tid & 31, wave = tid >> 5;
  const int s = blockIdx.x, t = blockIdx.y;
  const int base = (s * 32 + t) * 512;

  for (int idx = tid; idx < 512; idx += 128) (&xin[0][0])[idx] = xc[base + idx];
  __syncthreads();

  for (int r = wave; r < 16; r += 4) {
    float a0 = (r < 8) ? xin[r][lane] : 0.f;
    float a1 = (r < 8) ? xin[r][lane + 32] : 0.f;
    float s1 = a0 + a1, s2 = a0 * a0 + a1 * a1;
    float nz = (a0 != 0.f || a1 != 0.f) ? 1.f : 0.f;
#pragma unroll
    for (int off = 16; off > 0; off >>= 1) {
      s1 += __shfl_xor(s1, off, 32);
      s2 += __shfl_xor(s2, off, 32);
      nz += __shfl_xor(nz, off, 32);
    }
    float mk   = (r < 8 && nz > 0.f) ? 1.f : 0.f;
    float mean = s1 * (1.f / 64.f);
    float var  = s2 * (1.f / 64.f) - mean * mean;
    float inv  = rsqrtf(var + 1e-3f);
    xnh[r][lane]      = (f16)(((a0 - mean) * inv * ln_g[lane] + ln_b[lane]) * mk);
    xnh[r][lane + 32] = (f16)(((a1 - mean) * inv * ln_g[lane + 32] + ln_b[lane + 32]) * mk);
    if (lane == 0) maskrow[r] = mk;
  }
  __syncthreads();

  // t1 = gelu(xn@phi1.W1+b1), t2 = gelu(xn@phi2.W1+b1) : 16 tile-jobs
  for (int job = wave; job < 16; job += 4) {
    const bool first = job < 8;
    const int  nt    = first ? job : job - 8;
    const f16* Bw    = first ? wt_p1w1 : wt_p2w1;
    const float* bias = first ? p1b1 : p2b1;
    f16* dst = first ? &t1h[0][0] : &t2h[0][0];
    v8f c = {};
#pragma unroll
    for (int k0 = 0; k0 < 64; k0 += 32)
      c = wmma_f16(load_a_lds(&xnh[0][0], 64, k0, lane),
                   load_b_glb(Bw, 64, nt * 16, k0, lane), c);
    int ncol = nt * 16 + (lane & 15), mh = (lane >> 4) * 8;
    float bv = bias[ncol];
#pragma unroll
    for (int r = 0; r < 8; ++r) dst[(mh + r) * 128 + ncol] = (f16)gelu_f(c[r] + bv);
  }
  __syncthreads();

  // phi1 = (t1 @ phi1.W2 + b2) * mask
  for (int nt = wave; nt < 4; nt += 4) {
    v8f c = {};
#pragma unroll
    for (int k0 = 0; k0 < 128; k0 += 32)
      c = wmma_f16(load_a_lds(&t1h[0][0], 128, k0, lane),
                   load_b_glb(wt_p1w2, 128, nt * 16, k0, lane), c);
    int ncol = nt * 16 + (lane & 15), mh = (lane >> 4) * 8;
    float bv = p1b2[ncol];
#pragma unroll
    for (int r = 0; r < 8; ++r) {
      int m = mh + r;
      phi1f[m][ncol] = (c[r] + bv) * maskrow[m];
    }
  }
  // phi2 head (128 -> 2) per row, masked
  if (tid < 16) {
    int nrw = tid >> 1, hh = tid & 1;
    float acc = 0.f;
    for (int c = 0; c < 128; ++c)
      acc = __builtin_fmaf((float)t2h[nrw][c], p2w2[c * 2 + hh], acc);
    ph2[nrw][hh] = (acc + p2b2[hh]) * maskrow[nrw];
  }
  __syncthreads();

  // z[d,h] = sum_n phi1[n,d]*phi2[n,h]; x_rep[d] = gelu(z)@W3 + b3
  if (tid < 64) {
    int d = tid;
    float m0 = 0.f, m1 = 0.f;
#pragma unroll
    for (int n = 0; n < 8; ++n) {
      float p = phi1f[n][d];
      m0 = __builtin_fmaf(p, ph2[n][0], m0);
      m1 = __builtin_fmaf(p, ph2[n][1], m1);
    }
    xrep[(s * 32 + t) * 64 + d] = gelu_f(m0) * w3[0] + gelu_f(m1) * w3[1] + b3[0];
  }
}

// --------------------------------------------------------------- score ------
__global__ void score_kernel(const float* __restrict__ xrep,
                             const float* __restrict__ Wf,
                             const float* __restrict__ bf,
                             float* __restrict__ out) {
  int idx = blockIdx.x * blockDim.x + threadIdx.x;
  if (idx >= 1024) return;
  int i = idx >> 5, j = idx & 31;
  const float* a = xrep + idx * 64;
  const float* b = xrep + (j * 32 + i) * 64;
  float s = 0.f;
#pragma unroll 8
  for (int d = 0; d < 64; ++d) s = __builtin_fmaf(a[d], b[d], s);
  float v = (s * (1.f / 64.f)) * Wf[0] + bf[0];
  out[idx] = 1.f / (1.f + __expf(-v));   // v_exp_f32
}

// ------------------------------------------------------------- launcher -----
// Input order assumption: insertion order of setup_inputs() dict gives
// d_in[0]=x, d_in[1]=x_size, then params leaves in jax tree_flatten
// (alphabetically sorted nested dict keys) order.
extern "C" void kernel_launch(void* const* d_in, const int* in_sizes, int n_in,
                              void* d_out, int out_size, void* d_ws,
                              size_t ws_size, hipStream_t stream) {
  (void)in_sizes; (void)n_in; (void)out_size; (void)ws_size;
  const float* x     = (const float*)d_in[0];
  const float* W3f   = (const float*)d_in[2];
  const float* Wproj = (const float*)d_in[3];
  const float* Wf    = (const float*)d_in[4];
  const float* b3f   = (const float*)d_in[5];
  const float* bf    = (const float*)d_in[6];
  const float* c_W3  = (const float*)d_in[7];
  const float* c_b3  = (const float*)d_in[8];
  const float* c_lnb = (const float*)d_in[9];
  const float* c_lng = (const float*)d_in[10];
  const float* c_p1W1 = (const float*)d_in[11];
  const float* c_p1W2 = (const float*)d_in[12];
  const float* c_p1b1 = (const float*)d_in[13];
  const float* c_p1b2 = (const float*)d_in[14];
  const float* c_p2W1 = (const float*)d_in[15];
  const float* c_p2W2 = (const float*)d_in[16];
  const float* c_p2b1 = (const float*)d_in[17];
  const float* c_p2b2 = (const float*)d_in[18];
  const float* ln1b  = (const float*)d_in[19];
  const float* ln1g  = (const float*)d_in[20];
  const float* f_p1W1 = (const float*)d_in[21];
  const float* f_p1W2 = (const float*)d_in[22];
  const float* f_p1b1 = (const float*)d_in[23];
  const float* f_p1b2 = (const float*)d_in[24];
  const float* f_p2W1 = (const float*)d_in[25];
  const float* f_p2W2 = (const float*)d_in[26];
  const float* f_p2b1 = (const float*)d_in[27];
  const float* f_p2b2 = (const float*)d_in[28];
  const float* s_W3  = (const float*)d_in[29];
  const float* s_b3  = (const float*)d_in[30];
  const float* s_lnb = (const float*)d_in[31];
  const float* s_lng = (const float*)d_in[32];
  const float* s_p1W1 = (const float*)d_in[33];
  const float* s_p1W2 = (const float*)d_in[34];
  const float* s_p1b1 = (const float*)d_in[35];
  const float* s_p1b2 = (const float*)d_in[36];
  const float* s_p2W1 = (const float*)d_in[37];
  const float* s_p2W2 = (const float*)d_in[38];
  const float* s_p2b1 = (const float*)d_in[39];
  const float* s_p2b2 = (const float*)d_in[40];
  float* out = (float*)d_out;

  // workspace: f16 transposed weights, then f32 activation buffers
  f16* wt        = (f16*)d_ws;
  f16* wt_proj   = wt;                     // 64*64
  f16* wt_s_p1w1 = wt_proj + 64 * 64;      // [128][64]
  f16* wt_s_p1w2 = wt_s_p1w1 + 64 * 128;   // [64][128]
  f16* wt_s_p2w1 = wt_s_p1w2 + 128 * 64;   // [128][64]
  f16* wt_c_p1w1 = wt_s_p2w1 + 64 * 128;
  f16* wt_c_p1w2 = wt_c_p1w1 + 64 * 128;
  f16* wt_c_p2w1 = wt_c_p1w2 + 128 * 64;
  f16* wt_f_p1w1 = wt_c_p2w1 + 64 * 128;
  f16* wt_f_p1w2 = wt_f_p1w1 + 64 * 128;
  f16* wt_f_p2w1 = wt_f_p1w2 + 128 * 64;
  float* h_buf  = (float*)(wt + 81920);            // 32*8*64
  float* m1_buf = h_buf + 32 * 8 * 64;             // 32*8*64
  float* xc_buf = m1_buf + 32 * 8 * 64;            // 32*32*8*64
  float* xr_buf = xc_buf + 32 * 32 * 8 * 64;       // 32*32*64

  auto T = [&](const float* w, f16* dst, int K, int N) {
    int e = K * N;
    transpose_f16_kernel<<<dim3((e + 255) / 256), dim3(256), 0, stream>>>(w, dst, K, N);
  };
  T(Wproj, wt_proj, 64, 64);
  T(s_p1W1, wt_s_p1w1, 64, 128);
  T(s_p1W2, wt_s_p1w2, 128, 64);
  T(s_p2W1, wt_s_p2w1, 64, 128);
  T(c_p1W1, wt_c_p1w1, 64, 128);
  T(c_p1W2, wt_c_p1w2, 128, 64);
  T(c_p2W1, wt_c_p2w1, 64, 128);
  T(f_p1W1, wt_f_p1w1, 64, 128);
  T(f_p1W2, wt_f_p1w2, 128, 64);
  T(f_p2W1, wt_f_p2w1, 64, 128);

  proj_kernel<<<dim3(32), dim3(128), 0, stream>>>(x, wt_proj, h_buf);

  // self mixer output is independent of t -> compute once per s
  mixer_kernel<8, 8, false><<<dim3(32), dim3(128), 0, stream>>>(
      h_buf, s_lng, s_lnb, wt_s_p1w1, s_p1b1, wt_s_p1w2, s_p1b2, wt_s_p2w1,
      s_p2b1, s_p2W2, s_p2b2, s_W3, s_b3, m1_buf);

  mixer_kernel<16, 8, true><<<dim3(32, 32), dim3(128), 0, stream>>>(
      m1_buf, c_lng, c_lnb, wt_c_p1w1, c_p1b1, wt_c_p1w2, c_p1b2, wt_c_p2w1,
      c_p2b1, c_p2W2, c_p2b2, c_W3, c_b3, xc_buf);

  final_kernel<<<dim3(32, 32), dim3(128), 0, stream>>>(
      xc_buf, ln1g, ln1b, wt_f_p1w1, f_p1b1, wt_f_p1w2, f_p1b2, wt_f_p2w1,
      f_p2b1, f_p2W2, f_p2b2, W3f, b3f, xr_buf);

  score_kernel<<<dim3(4), dim3(256), 0, stream>>>(xr_buf, Wf, bf, out);
}
